// SpatialDecayAttention_73761768341563
// MI455X (gfx1250) — matrix operand order repaired and checked by reading, not compiled, not run on hardware
//
#include <hip/hip_runtime.h>
#include <math.h>

// ---------------------------------------------------------------------------
// Types for CDNA5 WMMA (gfx1250, wave32)
// ---------------------------------------------------------------------------
typedef __bf16 bf16;
typedef __attribute__((ext_vector_type(16))) __bf16 v16bf;
typedef __attribute__((ext_vector_type(4)))  __bf16 v4bf;
typedef __attribute__((ext_vector_type(8)))  float  v8f;
typedef __attribute__((ext_vector_type(4)))  float  v4f;
typedef __attribute__((ext_vector_type(4)))  int    v4i;

union FragB16 { v16bf v; v4i q[2]; };

__device__ __forceinline__ v4i ldg4(const bf16* p) {
  return *reinterpret_cast<const v4i*>(p);
}

__device__ __forceinline__ v8f wmma_bf16(v16bf a, v16bf b, v8f c) {
  return __builtin_amdgcn_wmma_f32_16x16x32_bf16(
      /*neg_a=*/false, a, /*neg_b=*/false, b,
      /*c_mod=*/(short)0, c, /*reuse_a=*/false, /*reuse_b=*/false);
}

__device__ __forceinline__ v8f zero8() {
  v8f z = {0.f, 0.f, 0.f, 0.f, 0.f, 0.f, 0.f, 0.f};
  return z;
}

// ---------------------------------------------------------------------------
// Problem constants (match reference)
// ---------------------------------------------------------------------------
#define BSZ   8
#define NTOK  1024     // 32*32
#define CDIM  512
#define NH    8
#define HD    64
#define GRIDW 32

// ---------------------------------------------------------------------------
// Kernel 0: f32 -> bf16 convert, 4 elements / thread (b128 load, b64 store)
// ---------------------------------------------------------------------------
__global__ void sda_cvt_bf16_x4(const float* __restrict__ in,
                                bf16* __restrict__ out, int n4) {
  int i = blockIdx.x * blockDim.x + threadIdx.x;
  if (i >= n4) return;
  v4f f = reinterpret_cast<const v4f*>(in)[i];
  v4bf o;
  o[0] = (bf16)f[0]; o[1] = (bf16)f[1];
  o[2] = (bf16)f[2]; o[3] = (bf16)f[3];
  reinterpret_cast<v4bf*>(out)[i] = o;
}

// ---------------------------------------------------------------------------
// Shared GEMM core: one wave computes a 32(row) x 64(col) tile of
//   y[n,d] = sum_c A[n,c] * W[d,c],  K = 512, software-pipelined fragments.
// Fragment layouts (v_wmma_f32_16x16x32_bf16, wave32):
//   A: lane L -> row L&15; 8 bf16 @ K k0=(L>>4)*8 and 8 bf16 @ k0+16
//   B: lane L -> col L&15; 8 bf16 @ K k0=(L>>4)*8 and 8 bf16 @ k0+16
//   C: lane L, vgpr r -> (row r+8*(L>>4), col L&15)
// ---------------------------------------------------------------------------
struct GemmTile {
  v8f acc[8];  // [rowHalf 0..1][colBlock 0..3] -> acc[rh*4+a]
};

__device__ __forceinline__ void gemm_32x64_k512(
    const bf16* __restrict__ arow0,   // A row for (row0 + lrow)
    const bf16* __restrict__ brow0,   // W row for (col0 + lrow)
    int kA0, int kB0, GemmTile& T) {
#pragma unroll
  for (int i = 0; i < 8; ++i) T.acc[i] = zero8();

  const bf16* arow1 = arow0 + (size_t)16 * CDIM;

  FragB16 A0c, A1c, Bc[4];
  A0c.q[0] = ldg4(arow0 + kA0);      A0c.q[1] = ldg4(arow0 + kA0 + 16);
  A1c.q[0] = ldg4(arow1 + kA0);      A1c.q[1] = ldg4(arow1 + kA0 + 16);
#pragma unroll
  for (int a = 0; a < 4; ++a) {
    const bf16* br = brow0 + (size_t)(a * 16) * CDIM;
    Bc[a].q[0] = ldg4(br + kB0);
    Bc[a].q[1] = ldg4(br + kB0 + 16);
  }

#pragma unroll
  for (int kb = 0; kb < CDIM; kb += 32) {
    FragB16 A0n, A1n, Bn[4];
    if (kb + 32 < CDIM) {
      int kn = kb + 32;
      __builtin_prefetch(arow0 + kn + 64, 0, 3);
      A0n.q[0] = ldg4(arow0 + kn + kA0);  A0n.q[1] = ldg4(arow0 + kn + kA0 + 16);
      A1n.q[0] = ldg4(arow1 + kn + kA0);  A1n.q[1] = ldg4(arow1 + kn + kA0 + 16);
#pragma unroll
      for (int a = 0; a < 4; ++a) {
        const bf16* br = brow0 + (size_t)(a * 16) * CDIM;
        Bn[a].q[0] = ldg4(br + kn + kB0);
        Bn[a].q[1] = ldg4(br + kn + kB0 + 16);
      }
    }
#pragma unroll
    for (int a = 0; a < 4; ++a) {
      T.acc[a]     = wmma_bf16(A0c.v, Bc[a].v, T.acc[a]);
      T.acc[4 + a] = wmma_bf16(A1c.v, Bc[a].v, T.acc[4 + a]);
    }
    if (kb + 32 < CDIM) {
      A0c = A0n; A1c = A1n;
#pragma unroll
      for (int a = 0; a < 4; ++a) Bc[a] = Bn[a];
    }
  }
}

// ---------------------------------------------------------------------------
// Kernel 1: QKV GEMM, scatter to Q [B,NH,N,HD], K [B,NH,N,HD], V^T [B,NH,HD,N]
// ---------------------------------------------------------------------------
__global__ __launch_bounds__(256) void sda_qkv_gemm(
    const bf16* __restrict__ xb, const bf16* __restrict__ wb,
    bf16* __restrict__ q, bf16* __restrict__ k, bf16* __restrict__ vt) {
  const int COLT = (3 * CDIM) / 64;  // 24 column tiles
  int wave = threadIdx.x >> 5;
  int lane = threadIdx.x & 31;
  int lrow = lane & 15;
  int hi   = lane >> 4;
  int t    = blockIdx.x * 8 + wave;            // 6144 wave tiles
  int rt   = t / COLT, ct = t % COLT;          // 256 row tiles x 24 col tiles
  int row0 = rt * 32, col0 = ct * 64;

  GemmTile T;
  gemm_32x64_k512(xb + (size_t)(row0 + lrow) * CDIM,
                  wb + (size_t)(col0 + lrow) * CDIM,
                  hi * 8, hi * 8, T);

  int sel = col0 >> 9;
  int h   = (col0 >> 6) & 7;
#pragma unroll
  for (int rh = 0; rh < 2; ++rh) {
#pragma unroll
    for (int a = 0; a < 4; ++a) {
      int e = a * 16 + lrow;
#pragma unroll
      for (int r = 0; r < 8; ++r) {
        int i  = row0 + rh * 16 + r + 8 * hi;   // token row in [0, 8192)
        int b  = i >> 10;
        int nn = i & (NTOK - 1);
        bf16 val = (bf16)T.acc[rh * 4 + a][r];
        size_t head = (size_t)(b * NH + h);
        if (sel == 0)       q [(head * NTOK + nn) * HD + e] = val;
        else if (sel == 1)  k [(head * NTOK + nn) * HD + e] = val;
        else                vt[(head * HD + e) * NTOK + nn] = val;
      }
    }
  }
}

// ---------------------------------------------------------------------------
// Kernel 2: flash-style attention with Manhattan-decay bias (exp2 domain).
// Grid: B*NH*(NTOK/128) blocks, 8 waves; each wave = 16 query rows x all keys.
// K fragments for j-block jb+32 are loaded before the softmax VALU section of
// block jb so exp2/shuffle work hides the load latency.
// ---------------------------------------------------------------------------
__global__ __launch_bounds__(256) void sda_attn(
    const bf16* __restrict__ q, const bf16* __restrict__ k,
    const bf16* __restrict__ vt, const float* __restrict__ log_decay,
    bf16* __restrict__ ao) {
  __shared__ __align__(16) bf16 plds[8][16 * 32];   // per-wave P tile (16x32)

  int wave = threadIdx.x >> 5;
  int lane = threadIdx.x & 31;
  int lrow = lane & 15;
  int hi   = lane >> 4;

  int blk  = blockIdx.x;          // 0..511
  int mblk = blk & 7;
  int bh   = blk >> 3;
  int h    = bh & 7;
  int b    = bh >> 3;
  int m0   = mblk * 128 + wave * 16;

  const size_t base  = (size_t)(b * NH + h) * NTOK * HD;  // q,k
  const size_t vbase = (size_t)(b * NH + h) * HD * NTOK;  // vt

  const float L2E = 1.4426950408889634f;
  float decay = log1pf(expf(log_decay[h]));   // softplus
  float dl2   = decay * L2E;
  float sl2   = 0.125f * L2E;                 // hd^-0.5 * log2(e)

  const int kA0 = hi * 8;
  const int kB0 = hi * 8;

  // Persistent Q fragments (16 rows x 64 head-dim -> two K=32 fragments)
  const bf16* qrow = q + base + (size_t)(m0 + lrow) * HD;
  FragB16 Aq0, Aq1;
  Aq0.q[0] = ldg4(qrow + kA0);       Aq0.q[1] = ldg4(qrow + kA0 + 16);
  Aq1.q[0] = ldg4(qrow + 32 + kA0);  Aq1.q[1] = ldg4(qrow + 32 + kA0 + 16);

  float mrow[8], lsum[8];
  int ri[8], ci[8];
#pragma unroll
  for (int r = 0; r < 8; ++r) {
    mrow[r] = -3.0e38f;
    lsum[r] = 0.f;
    int i = m0 + r + 8 * hi;
    ri[r] = i >> 5;
    ci[r] = i & (GRIDW - 1);
  }

  v8f acc[4];
  acc[0] = zero8(); acc[1] = zero8(); acc[2] = zero8(); acc[3] = zero8();

  // Prime K fragments for jb = 0.
  FragB16 B0l, B0h, B1l, B1h;
  {
    const bf16* kr0 = k + base + (size_t)lrow * HD;
    const bf16* kr1 = kr0 + 16 * HD;
    B0l.q[0] = ldg4(kr0 + kB0);      B0l.q[1] = ldg4(kr0 + kB0 + 16);
    B0h.q[0] = ldg4(kr0 + 32 + kB0); B0h.q[1] = ldg4(kr0 + 32 + kB0 + 16);
    B1l.q[0] = ldg4(kr1 + kB0);      B1l.q[1] = ldg4(kr1 + kB0 + 16);
    B1h.q[0] = ldg4(kr1 + 32 + kB0); B1h.q[1] = ldg4(kr1 + 32 + kB0 + 16);
  }

  for (int jb = 0; jb < NTOK; jb += 32) {
    // ---- S = Q @ K^T for columns jb..jb+31 (two 16x16 tiles) ----
    v8f s0 = zero8();
    s0 = wmma_bf16(Aq0.v, B0l.v, s0);
    s0 = wmma_bf16(Aq1.v, B0h.v, s0);
    v8f s1 = zero8();
    s1 = wmma_bf16(Aq0.v, B1l.v, s1);
    s1 = wmma_bf16(Aq1.v, B1h.v, s1);

    // ---- pipeline: K fragments for jb+32 ----
    if (jb + 32 < NTOK) {
      const bf16* kr0 = k + base + (size_t)(jb + 32 + lrow) * HD;
      const bf16* kr1 = kr0 + 16 * HD;
      B0l.q[0] = ldg4(kr0 + kB0);      B0l.q[1] = ldg4(kr0 + kB0 + 16);
      B0h.q[0] = ldg4(kr0 + 32 + kB0); B0h.q[1] = ldg4(kr0 + 32 + kB0 + 16);
      B1l.q[0] = ldg4(kr1 + kB0);      B1l.q[1] = ldg4(kr1 + kB0 + 16);
      B1h.q[0] = ldg4(kr1 + 32 + kB0); B1h.q[1] = ldg4(kr1 + 32 + kB0 + 16);
    }

    // ---- pipeline: V fragments for this jb (consumed after softmax) ----
    FragB16 Bv[4];
    {
      const bf16* vr = vt + vbase + (size_t)lrow * NTOK + jb;
#pragma unroll
      for (int a = 0; a < 4; ++a) {
        const bf16* vp = vr + (size_t)(a * 16) * NTOK;
        Bv[a].q[0] = ldg4(vp + kB0);
        Bv[a].q[1] = ldg4(vp + kB0 + 16);
      }
    }

    // ---- bias + online softmax (exp2 domain) ----
    int j0 = jb + lrow, j1 = j0 + 16;
    int rj0 = j0 >> 5, cj0 = j0 & (GRIDW - 1);
    int rj1 = j1 >> 5, cj1 = j1 & (GRIDW - 1);

#pragma unroll
    for (int r = 0; r < 8; ++r) {
      int dr0 = ri[r] - rj0; dr0 = dr0 < 0 ? -dr0 : dr0;
      int dc0 = ci[r] - cj0; dc0 = dc0 < 0 ? -dc0 : dc0;
      int dr1 = ri[r] - rj1; dr1 = dr1 < 0 ? -dr1 : dr1;
      int dc1 = ci[r] - cj1; dc1 = dc1 < 0 ? -dc1 : dc1;
      float t0 = s0[r] * sl2 - dl2 * (float)(dr0 + dc0);
      float t1 = s1[r] * sl2 - dl2 * (float)(dr1 + dc1);

      float rm = fmaxf(t0, t1);
#pragma unroll
      for (int msk = 1; msk < 16; msk <<= 1)
        rm = fmaxf(rm, __shfl_xor(rm, msk, 32));
      float mnew  = fmaxf(mrow[r], rm);
      float alpha = exp2f(mrow[r] - mnew);
      float p0 = exp2f(t0 - mnew);
      float p1 = exp2f(t1 - mnew);
      float rs = p0 + p1;
#pragma unroll
      for (int msk = 1; msk < 16; msk <<= 1)
        rs += __shfl_xor(rs, msk, 32);
      lsum[r] = lsum[r] * alpha + rs;
      mrow[r] = mnew;
      acc[0][r] *= alpha; acc[1][r] *= alpha;
      acc[2][r] *= alpha; acc[3][r] *= alpha;

      int prow = r + 8 * hi;
      plds[wave][prow * 32 + lrow]      = (bf16)p0;   // ds_store_b16
      plds[wave][prow * 32 + lrow + 16] = (bf16)p1;
    }

    // ---- reshape P (C-layout -> A-layout) via wave-private LDS ----
    // DS ops from the same wave complete in order: no barrier needed.
    FragB16 Ap;
    Ap.q[0] = *reinterpret_cast<const v4i*>(&plds[wave][lrow * 32 + kA0]);
    Ap.q[1] = *reinterpret_cast<const v4i*>(&plds[wave][lrow * 32 + kA0 + 16]);

    // ---- acc += P @ V ----
#pragma unroll
    for (int a = 0; a < 4; ++a)
      acc[a] = wmma_bf16(Ap.v, Bv[a].v, acc[a]);
  }

  // ---- finalize: divide by row sum, write [B,N,C] with head offset ----
#pragma unroll
  for (int r = 0; r < 8; ++r) {
    float inv = 1.0f / lsum[r];
    int i = m0 + r + 8 * hi;
    size_t orow = ((size_t)b * NTOK + i) * CDIM + h * HD;
#pragma unroll
    for (int a = 0; a < 4; ++a)
      ao[orow + a * 16 + lrow] = (bf16)(acc[a][r] * inv);
  }
}

// ---------------------------------------------------------------------------
// Kernel 3: output projection  out[n,d] = sum_c ao[n,c] * wproj[d,c]  (f32 out)
// ---------------------------------------------------------------------------
__global__ __launch_bounds__(256) void sda_proj_gemm(
    const bf16* __restrict__ ab, const bf16* __restrict__ wb,
    float* __restrict__ out) {
  const int COLT = CDIM / 64;  // 8 column tiles
  int wave = threadIdx.x >> 5;
  int lane = threadIdx.x & 31;
  int lrow = lane & 15;
  int hi   = lane >> 4;
  int t    = blockIdx.x * 8 + wave;            // 2048 wave tiles
  int rt   = t / COLT, ct = t % COLT;          // 256 row tiles x 8 col tiles
  int row0 = rt * 32, col0 = ct * 64;

  GemmTile T;
  gemm_32x64_k512(ab + (size_t)(row0 + lrow) * CDIM,
                  wb + (size_t)(col0 + lrow) * CDIM,
                  hi * 8, hi * 8, T);

#pragma unroll
  for (int rh = 0; rh < 2; ++rh) {
#pragma unroll
    for (int r = 0; r < 8; ++r) {
      int i = row0 + rh * 16 + r + 8 * hi;
      float* orow = out + (size_t)i * CDIM + col0;
#pragma unroll
      for (int a = 0; a < 4; ++a)
        orow[a * 16 + lrow] = T.acc[rh * 4 + a][r];
    }
  }
}

// ---------------------------------------------------------------------------
// Host launcher
// ---------------------------------------------------------------------------
extern "C" void kernel_launch(void* const* d_in, const int* in_sizes, int n_in,
                              void* d_out, int out_size, void* d_ws, size_t ws_size,
                              hipStream_t stream) {
  (void)in_sizes; (void)n_in; (void)out_size; (void)ws_size;
  const float* x     = (const float*)d_in[0];  // [8,1024,512]
  const float* wqkv  = (const float*)d_in[1];  // [1536,512]
  const float* wproj = (const float*)d_in[2];  // [512,512]
  const float* ldk   = (const float*)d_in[3];  // [8]

  const size_t NX  = (size_t)BSZ * NTOK * CDIM;       // 4,194,304
  const size_t NWQ = (size_t)3 * CDIM * CDIM;         //   786,432
  const size_t NWP = (size_t)CDIM * CDIM;             //   262,144
  const size_t NQK = (size_t)BSZ * NH * NTOK * HD;    // 4,194,304

  char* ws = (char*)d_ws;
  size_t off = 0;
  auto carve = [&](size_t bytes) -> void* {
    void* p = ws + off;
    off += (bytes + 255) & ~(size_t)255;
    return p;
  };
  bf16* xb  = (bf16*)carve(NX  * 2);
  bf16* wqb = (bf16*)carve(NWQ * 2);
  bf16* wpb = (bf16*)carve(NWP * 2);
  bf16* qb  = (bf16*)carve(NQK * 2);
  bf16* kb  = (bf16*)carve(NQK * 2);
  bf16* vtb = (bf16*)carve(NQK * 2);
  bf16* aob = (bf16*)carve(NX  * 2);

  // 0) converts
  sda_cvt_bf16_x4<<<(int)(NX  / 4 / 256), 256, 0, stream>>>(x,     xb,  (int)(NX  / 4));
  sda_cvt_bf16_x4<<<(int)(NWQ / 4 / 256), 256, 0, stream>>>(wqkv,  wqb, (int)(NWQ / 4));
  sda_cvt_bf16_x4<<<(int)(NWP / 4 / 256), 256, 0, stream>>>(wproj, wpb, (int)(NWP / 4));

  // 1) QKV gemm: (8192/32) * (1536/64) = 6144 wave tiles / 8 waves = 768 blocks
  sda_qkv_gemm<<<768, 256, 0, stream>>>(xb, wqb, qb, kb, vtb);

  // 2) attention: B*NH*(N/128) = 512 blocks
  sda_attn<<<512, 256, 0, stream>>>(qb, kb, vtb, ldk, aob);

  // 3) projection: (8192/32) * (512/64) = 2048 wave tiles / 8 = 256 blocks
  sda_proj_gemm<<<256, 256, 0, stream>>>(aob, wpb, (float*)d_out);
}
